// MultiHeadSelfAttention_76665166233808
// MI455X (gfx1250) — matrix operand order repaired and checked by reading, not compiled, not run on hardware
//
#include <hip/hip_runtime.h>
#include <stdint.h>

// ---------------- problem constants ----------------
static constexpr int kBatch = 8;
static constexpr int kNMask = 1024;
static constexpr int kNTok  = 1025;          // NMASK + 1
static constexpr int kDim   = 768;
static constexpr int kHeads = 12;
static constexpr int kHDim  = 64;
static constexpr int kNPad  = 1056;          // 33 * 32, padded token count
static constexpr int kQkvCols = 3 * kDim;    // 2304
static constexpr float kScale = 0.125f;      // 64^-0.5
static constexpr float kNegMask = -1.0e9f;   // reference's finite -inf
static constexpr float kNegPad  = -3.0e38f;  // hard kill for pad keys

typedef __bf16 bf16;
typedef __attribute__((ext_vector_type(16))) __bf16 v16bf;
typedef __attribute__((ext_vector_type(8)))  float  v8f;

// D = A(16x32 bf16) * B(32x16 bf16) + C(16x16 f32)
static __device__ __forceinline__ v8f wmma_bf16(v16bf a, v16bf b, v8f c) {
  return __builtin_amdgcn_wmma_f32_16x16x32_bf16(
      /*neg_a=*/false, a, /*neg_b=*/false, b,
      /*c_mod=*/(short)0, c, /*reuse_a=*/false, /*reuse_b=*/false);
}

// ---- fragment loaders -------------------------------------------------
// A fragment (16x32, bf16): lane (h = lane>>4, m = lane&15) holds
// element e -> K = kk + 8*h + e + 8*(e>=8)   (ISA 7.12.2, 16-bit A layout)
// rowk = &base[m*ld + kk]; two 16B vector loads (elements 0..7 / 8..15)
static __device__ __forceinline__ v16bf load_a_bf16(const bf16* rowk, int h) {
  union { v16bf v; uint4 q[2]; } u;
  u.q[0] = *(const uint4*)(rowk + 8 * h);
  u.q[1] = *(const uint4*)(rowk + 16 + 8 * h);
  return u.v;
}

// B fragment (32x16, bf16): lane holds column n = lane&15,
// element e -> K = 8*h + e + 8*(e>=8) (same as A); p = &base[k0*ld + n].
static __device__ __forceinline__ v16bf load_b_bf16(const bf16* p, int ld, int h) {
  v16bf v;
#pragma unroll
  for (int e = 0; e < 16; ++e) v[e] = p[(size_t)(8 * h + e + 8 * (e >= 8)) * ld];
  return v;
}

// ---------------- conversion kernels ----------------------------------
// f32 -> bf16, 4 elements/thread (n must be a multiple of 4)
__global__ void cvt_f32_bf16_kernel(const float* __restrict__ src,
                                    bf16* __restrict__ dst, int n4) {
  int i = blockIdx.x * blockDim.x + threadIdx.x;
  if (i >= n4) return;
  float4 f = ((const float4*)src)[i];
  bf16* d = dst + (size_t)i * 4;
  d[0] = (bf16)f.x; d[1] = (bf16)f.y; d[2] = (bf16)f.z; d[3] = (bf16)f.w;
}

// x (B, 1025, 768) f32 -> Xb (B, kNPad, 768) bf16, zero pad rows
__global__ void cvt_pad_x_kernel(const float* __restrict__ x,
                                 bf16* __restrict__ Xb) {
  int i = blockIdx.x * blockDim.x + threadIdx.x;        // one thread / 4 elems
  const int row4 = kDim / 4;                            // 192
  int total = kBatch * kNPad * row4;
  if (i >= total) return;
  int d4 = i % row4;
  int n  = (i / row4) % kNPad;
  int b  = i / (row4 * kNPad);
  float4 f = make_float4(0.f, 0.f, 0.f, 0.f);
  if (n < kNTok)
    f = ((const float4*)(x + ((size_t)b * kNTok + n) * kDim))[d4];
  bf16* d = Xb + (((size_t)b * kNPad + n) * kDim) + d4 * 4;
  d[0] = (bf16)f.x; d[1] = (bf16)f.y; d[2] = (bf16)f.z; d[3] = (bf16)f.w;
}

// ---------------- kernel 1: QKV projection ----------------------------
// one wave -> 32 (tokens) x 64 (qkv cols); writes Q/KT/V bf16 layouts
__global__ void qkv_gemm_kernel(const bf16* __restrict__ Xb,
                                const bf16* __restrict__ Wb,
                                const float* __restrict__ bias,
                                bf16* __restrict__ Qb,
                                bf16* __restrict__ KT,
                                bf16* __restrict__ Vb) {
  const int MT = kNPad / 32;      // 33
  const int CT = kQkvCols / 64;   // 36
  int wid  = (blockIdx.x * blockDim.x + threadIdx.x) >> 5;
  int lane = threadIdx.x & 31;
  if (wid >= kBatch * MT * CT) return;
  int ct = wid % CT;
  int mt = (wid / CT) % MT;
  int b  = wid / (CT * MT);
  int hl = lane >> 4, li = lane & 15;

  int m0 = mt * 32;
  const bf16* xrow0 = Xb + ((size_t)b * kNPad + m0 + li) * kDim;
  const bf16* xrow1 = xrow0 + (size_t)16 * kDim;
  int c0 = ct * 64;

  v8f acc[2][4];
#pragma unroll
  for (int f = 0; f < 4; ++f) {
    float bv = bias[c0 + 16 * f + li];
#pragma unroll
    for (int r = 0; r < 8; ++r) { acc[0][f][r] = bv; acc[1][f][r] = bv; }
  }

  int krow = li + 16 * hl;
  for (int kk = 0; kk < kDim; kk += 32) {
    v16bf a0 = load_a_bf16(xrow0 + kk, hl);
    v16bf a1 = load_a_bf16(xrow1 + kk, hl);
    const bf16* wp = Wb + (size_t)kk * kQkvCols + c0 + li;
#pragma unroll
    for (int f = 0; f < 4; ++f) {
      v16bf bm = load_b_bf16(wp + 16 * f, kQkvCols, hl);
      acc[0][f] = wmma_bf16(a0, bm, acc[0][f]);
      acc[1][f] = wmma_bf16(a1, bm, acc[1][f]);
    }
  }

  // scatter: column decides q/k/v + head + dim (constant per lane,f)
#pragma unroll
  for (int f = 0; f < 4; ++f) {
    int c = c0 + 16 * f + li;
    int which = c / kDim;
    int rem = c - which * kDim;
    int hh = rem >> 6;
    int d  = rem & 63;
#pragma unroll
    for (int g = 0; g < 2; ++g) {
#pragma unroll
      for (int r = 0; r < 8; ++r) {
        int n = m0 + 16 * g + r + 8 * hl;     // C row -> token (may be pad)
        float v = acc[g][f][r];
        if (which == 0)
          Qb[(((size_t)b * kHeads + hh) * kNPad + n) * kHDim + d] = (bf16)(v * kScale);
        else if (which == 1)
          KT[(((size_t)b * kHeads + hh) * kHDim + d) * kNPad + n] = (bf16)v;
        else
          Vb[(((size_t)b * kHeads + hh) * kNPad + n) * kHDim + d] = (bf16)v;
      }
    }
  }
}

// ---------------- kernel 2: flash attention ---------------------------
// one wave -> (b, h, 16 query rows); streams keys 32 at a time
__global__ void attn_kernel(const bf16* __restrict__ Qb,
                            const bf16* __restrict__ KT,
                            const bf16* __restrict__ Vb,
                            const unsigned char* __restrict__ mask,
                            bf16* __restrict__ AO) {
  __shared__ __attribute__((aligned(16))) bf16 pbuf[8][16 * 32]; // 1KB/wave
  const int QT = kNPad / 16;  // 66
  int widInBlk = threadIdx.x >> 5;
  int wid  = (blockIdx.x * blockDim.x + threadIdx.x) >> 5;
  int lane = threadIdx.x & 31;
  if (wid >= kBatch * kHeads * QT) return;
  int qt = wid % QT;
  int h  = (wid / QT) % kHeads;
  int b  = wid / (QT * kHeads);
  int hl = lane >> 4, li = lane & 15;
  int n0 = qt * 16;

  const bf16* Qbh  = Qb + (((size_t)b * kHeads + h) * kNPad) * kHDim;
  const bf16* KTbh = KT + (((size_t)b * kHeads + h) * kHDim) * kNPad;
  const bf16* Vbh  = Vb + (((size_t)b * kHeads + h) * kNPad) * kHDim;
  const unsigned char* mrow = mask + (size_t)b * kNMask;

  // Q fragments, loaded once (head-dim 64 = 2 x K32)
  const bf16* qrow = Qbh + (size_t)(n0 + li) * kHDim;
  v16bf qa0 = load_a_bf16(qrow + 0, hl);
  v16bf qa1 = load_a_bf16(qrow + 32, hl);

  // query validity per C row (row = r + 8*hl)
  bool qv[8];
#pragma unroll
  for (int r = 0; r < 8; ++r) {
    int i = n0 + r + 8 * hl;
    qv[r] = (i == 0) || (i < kNTok ? (mrow[i - 1] != 0) : true);
  }

  v8f o0 = {}, o1 = {}, o2 = {}, o3 = {};
  float mx[8], ls[8];
#pragma unroll
  for (int r = 0; r < 8; ++r) { mx[r] = -3.0e38f; ls[r] = 0.0f; }

  bf16* wb = pbuf[widInBlk];
  int krow = li + 16 * hl;

  for (int j0 = 0; j0 < kNPad; j0 += 32) {
    // S = Q * K^T  (two 16x16 logit tiles)
    const bf16* kp0  = KTbh + j0 + li;
    const bf16* kp32 = KTbh + (size_t)32 * kNPad + j0 + li;
    v8f s0 = {}, s1 = {};
    s0 = wmma_bf16(qa0, load_b_bf16(kp0, kNPad, hl), s0);
    s0 = wmma_bf16(qa1, load_b_bf16(kp32, kNPad, hl), s0);
    s1 = wmma_bf16(qa0, load_b_bf16(kp0 + 16, kNPad, hl), s1);
    s1 = wmma_bf16(qa1, load_b_bf16(kp32 + 16, kNPad, hl), s1);

    // key validity: each lane owns columns jA (tile0) and jB (tile1)
    int jA = j0 + li, jB = jA + 16;
    bool padA = jA >= kNTok, padB = jB >= kNTok;
    bool kvA = !padA && (jA == 0 || mrow[jA - 1] != 0);
    bool kvB = !padB && (jB == 0 || mrow[jB - 1] != 0);

#pragma unroll
    for (int r = 0; r < 8; ++r) {
      float lA = padA ? kNegPad : ((kvA && qv[r]) ? s0[r] : kNegMask);
      float lB = padB ? kNegPad : ((kvB && qv[r]) ? s1[r] : kNegMask);
      // row-max across the 16 lanes of this half (row lives across lanes)
      float t = fmaxf(lA, lB);
#pragma unroll
      for (int d = 1; d < 16; d <<= 1) t = fmaxf(t, __shfl_xor(t, d, 32));
      float mnew = fmaxf(mx[r], t);
      float corr = __expf(mx[r] - mnew);
      mx[r] = mnew;
      float pA = __expf(lA - mnew);
      float pB = __expf(lB - mnew);
      float rs = pA + pB;
#pragma unroll
      for (int d = 1; d < 16; d <<= 1) rs += __shfl_xor(rs, d, 32);
      ls[r] = ls[r] * corr + rs;
      o0[r] *= corr; o1[r] *= corr; o2[r] *= corr; o3[r] *= corr;
      // stage P (bf16) into LDS for the C->A transpose
      int rr = r + 8 * hl;
      wb[rr * 32 + li]      = (bf16)pA;
      wb[rr * 32 + li + 16] = (bf16)pB;
    }
    __asm__ volatile("s_wait_dscnt 0" ::: "memory");  // same-wave LDS order
    v16bf pa = load_a_bf16(wb + (size_t)li * 32, hl);
    const bf16* vp = Vbh + (size_t)j0 * kHDim + li;
    o0 = wmma_bf16(pa, load_b_bf16(vp + 0, kHDim, hl),  o0);
    o1 = wmma_bf16(pa, load_b_bf16(vp + 16, kHDim, hl), o1);
    o2 = wmma_bf16(pa, load_b_bf16(vp + 32, kHDim, hl), o2);
    o3 = wmma_bf16(pa, load_b_bf16(vp + 48, kHDim, hl), o3);
  }

  // normalize + store attn output (bf16) for projection GEMM
#pragma unroll
  for (int r = 0; r < 8; ++r) {
    float inv = 1.0f / ls[r];
    int i = n0 + r + 8 * hl;
    bf16* op = AO + ((size_t)b * kNPad + i) * kDim + h * kHDim;
    op[li +  0] = (bf16)(o0[r] * inv);
    op[li + 16] = (bf16)(o1[r] * inv);
    op[li + 32] = (bf16)(o2[r] * inv);
    op[li + 48] = (bf16)(o3[r] * inv);
  }
}

// ---------------- kernel 3: output projection -------------------------
// one wave -> 32 rows x 64 cols, bf16 A and bf16 W, f32 output
__global__ void proj_gemm_kernel(const bf16* __restrict__ AO,
                                 const bf16* __restrict__ Wb,
                                 const float* __restrict__ bias,
                                 float* __restrict__ out) {
  const int MT = kNPad / 32;     // 33 (1056 rows cover 1025)
  const int CT = kDim / 64;      // 12
  int wid  = (blockIdx.x * blockDim.x + threadIdx.x) >> 5;
  int lane = threadIdx.x & 31;
  if (wid >= kBatch * MT * CT) return;
  int ct = wid % CT;
  int mt = (wid / CT) % MT;
  int b  = wid / (CT * MT);
  int hl = lane >> 4, li = lane & 15;

  int m0 = mt * 32;
  const bf16* arow0 = AO + ((size_t)b * kNPad + m0 + li) * kDim;
  const bf16* arow1 = arow0 + (size_t)16 * kDim;
  int c0 = ct * 64;

  v8f acc[2][4];
#pragma unroll
  for (int f = 0; f < 4; ++f) {
    float bv = bias[c0 + 16 * f + li];
#pragma unroll
    for (int r = 0; r < 8; ++r) { acc[0][f][r] = bv; acc[1][f][r] = bv; }
  }

  int krow = li + 16 * hl;
  for (int kk = 0; kk < kDim; kk += 32) {
    v16bf a0 = load_a_bf16(arow0 + kk, hl);
    v16bf a1 = load_a_bf16(arow1 + kk, hl);
    const bf16* wp = Wb + (size_t)kk * kDim + c0 + li;
#pragma unroll
    for (int f = 0; f < 4; ++f) {
      v16bf bm = load_b_bf16(wp + 16 * f, kDim, hl);
      acc[0][f] = wmma_bf16(a0, bm, acc[0][f]);
      acc[1][f] = wmma_bf16(a1, bm, acc[1][f]);
    }
  }

#pragma unroll
  for (int f = 0; f < 4; ++f) {
#pragma unroll
    for (int g = 0; g < 2; ++g) {
#pragma unroll
      for (int r = 0; r < 8; ++r) {
        int n = m0 + 16 * g + r + 8 * hl;
        if (n < kNTok)
          out[((size_t)b * kNTok + n) * kDim + c0 + 16 * f + li] = acc[g][f][r];
      }
    }
  }
}

// ---------------- launcher --------------------------------------------
extern "C" void kernel_launch(void* const* d_in, const int* in_sizes, int n_in,
                              void* d_out, int out_size, void* d_ws, size_t ws_size,
                              hipStream_t stream) {
  const float* x      = (const float*)d_in[0];
  const unsigned char* mask = (const unsigned char*)d_in[1];  // jax bool = 1B
  const float* w_qkv  = (const float*)d_in[2];
  const float* b_qkv  = (const float*)d_in[3];
  const float* w_proj = (const float*)d_in[4];
  const float* b_proj = (const float*)d_in[5];
  float* out = (float*)d_out;

  // workspace layout (bf16 elements):
  //   Xb | Wqkv | Wproj | Q | KT | V | AO     (~70MB total, L2-resident)
  size_t per   = (size_t)kBatch * kHeads * kNPad * kHDim;  // == kBatch*kNPad*kDim
  size_t nWqkv = (size_t)kDim * kQkvCols;                  // 1,769,472
  size_t nWprj = (size_t)kDim * kDim;                      //   589,824
  bf16* Xb  = (bf16*)d_ws;
  bf16* Wq  = Xb + per;
  bf16* Wp  = Wq + nWqkv;
  bf16* Qb  = Wp + nWprj;
  bf16* KT  = Qb + per;
  bf16* Vb  = KT + per;
  bf16* AO  = Vb + per;

  // --- precision/layout prep (one-time converts, removes cvt from hot loops)
  {
    int n4 = (int)(nWqkv / 4);
    cvt_f32_bf16_kernel<<<(n4 + 255) / 256, 256, 0, stream>>>(w_qkv, Wq, n4);
  }
  {
    int n4 = (int)(nWprj / 4);
    cvt_f32_bf16_kernel<<<(n4 + 255) / 256, 256, 0, stream>>>(w_proj, Wp, n4);
  }
  {
    int total = kBatch * kNPad * (kDim / 4);
    cvt_pad_x_kernel<<<(total + 255) / 256, 256, 0, stream>>>(x, Xb);
  }

  // --- QKV projection
  {
    int waves  = kBatch * (kNPad / 32) * (kQkvCols / 64);  // 9504
    int blocks = (waves + 7) / 8;
    qkv_gemm_kernel<<<blocks, 256, 0, stream>>>(Xb, Wq, b_qkv, Qb, KT, Vb);
  }
  // --- fused masked softmax attention
  {
    int waves  = kBatch * kHeads * (kNPad / 16);           // 6336
    int blocks = (waves + 7) / 8;
    attn_kernel<<<blocks, 256, 0, stream>>>(Qb, KT, Vb, mask, AO);
  }
  // --- output projection
  {
    int waves  = kBatch * (kNPad / 32) * (kDim / 64);      // 3168
    int blocks = (waves + 7) / 8;
    proj_gemm_kernel<<<blocks, 256, 0, stream>>>(AO, Wp, b_proj, out);
  }
}
